// CritiGraph_66391604462091
// MI455X (gfx1250) — compile-verified, matrix operand order checked by reading, not run on hardware
//
#include <hip/hip_runtime.h>
#include <hip/hip_bf16.h>

#define H_ 15
#define K_ 15
#define TP 8
#define NNODES 100000
#define BATCH 128
#define DMAX 64
#define NCAND 226          // H*K + 1
#define NTILE 15           // ceil(NCAND/16)
#define EPSV 0.1f
#define LN2V 0.69314718055994530942f

typedef __attribute__((ext_vector_type(16))) _Float16 v16h;
typedef __attribute__((ext_vector_type(8)))  float    v8f;

// lut[x] = 0 if x==0 else floor(log2(x))+1  ==  32 - clz(x)
__device__ __forceinline__ float lutdist(int x) {
    return x ? (float)(32 - __clz(x)) : 0.0f;
}
// hardware v_rcp_f32 / v_log_f32 (inputs here are >= 0.1: no denorm fixups needed)
__device__ __forceinline__ float fastrcp(float x)  { return __builtin_amdgcn_rcpf(x); }
__device__ __forceinline__ float fastlog2(float x) { return __builtin_amdgcn_logf(x); }

__global__ __launch_bounds__(256) void critigraph_main(
    const int* __restrict__ sta_ind, const int* __restrict__ locations,
    const int* __restrict__ degree,  const int* __restrict__ pos_ind,
    const int* __restrict__ neg_ind, const int* __restrict__ rmask,
    const int* __restrict__ perm,
    float* __restrict__ ws_loss, int* __restrict__ ws_sel)
{
    __shared__ int   s_sta[TP];
    __shared__ int   s_pos[DMAX][TP];
    __shared__ int   s_neg[DMAX][TP];
    __shared__ float s_offp[DMAX][TP];   // per-(d,t) fused offset: aa = fma(dc, sp, offp)
    __shared__ float s_offn[DMAX][TP];
    __shared__ float s_sp[DMAX];         // 0.125*rlp (0 for masked d)
    __shared__ float s_sn[DMAX];

    const int b    = blockIdx.x;
    const int tid  = threadIdx.x;
    const int lane = tid & 31;
    const int wave = tid >> 5;     // = coordinate t (8 waves of wave32)

    const int sta = sta_ind[b];
    if (tid < TP) s_sta[tid] = locations[sta * TP + tid];
    __syncthreads();

    // ---- per-d precompute (64 threads) ----
    int validp = 0;
    if (tid < DMAX) {
        const int d    = tid;
        const int pind = pos_ind[b * DMAX + d];
        const int nind = neg_ind[b * DMAX + d];
        validp = (pind >= 0);
        const int pw = pind < 0 ? pind + NNODES : pind;
        const int nw = nind < 0 ? nind + NNODES : nind;
        const int deg1  = degree[sta];
        const float rlp = fastrcp(__logf((float)((deg1 + 1) * (degree[pw] + 1))));
        const float rln = fastrcp(__logf((float)((deg1 + 1) * (degree[nw] + 1))));
        if (validp) {
            float dp[TP], dn[TP];
            float ps = 0.0f, ns = 0.0f;
            for (int t = 0; t < TP; ++t) {
                const int pl = locations[pind * TP + t];
                const int nl = locations[nind * TP + t];
                s_pos[d][t] = pl;  s_neg[d][t] = nl;
                dp[t] = lutdist(s_sta[t] ^ pl);
                dn[t] = lutdist(s_sta[t] ^ nl);
                ps += dp[t];  ns += dn[t];
            }
            const float sp = 0.125f * rlp, sn = 0.125f * rln;
            s_sp[d] = sp;  s_sn[d] = sn;
            for (int t = 0; t < TP; ++t) {
                // aa = ((dc - dsp + psum)/8 + EPS)*rlp == fma(dc, sp, offp)
                s_offp[d][t] = (ps - dp[t]) * sp + EPSV * rlp;
                s_offn[d][t] = (ns - dn[t]) * sn + EPSV * rln;
            }
        } else {
            // masked d: dis_new = 0 (pos) / 1000 (neg) -> constant aa, constant
            // contribution; scale=0 makes the hot loop produce it exactly, so the
            // WMMA sum includes the masked terms for free (no Cpos/Cneg).
            s_sp[d] = 0.0f;  s_sn[d] = 0.0f;
            for (int t = 0; t < TP; ++t) {
                s_pos[d][t] = 0;  s_neg[d][t] = 0;
                s_offp[d][t] = EPSV * rlp;
                s_offn[d][t] = (1000.0f + EPSV) * rln;
            }
        }
    }
    const int lgcnt = __syncthreads_count(validp);       // = lg, also barriers LDS
    const float scale = LN2V * fastrcp((float)lgcnt);    // ln2 / lg (log2 -> ln fold)

    // ---- per-wave candidate scoring: wave handles coordinate t = wave ----
    const int t      = wave;
    const int m      = lane & 15;
    const int hi     = lane >> 4;          // K half-select per A layout
    const int staloc = s_sta[t];

    v16h bOnes;
    #pragma unroll
    for (int e = 0; e < 16; ++e) bOnes[e] = (_Float16)1.0f;

    float bestTot = 1e30f, bestPos = 0.0f, bestNeg = 0.0f;
    int   bestIdx = 0;

    for (int tile = 0; tile < NTILE; ++tile) {
        // candidate location for this lane's A-matrix row (M = lane&15)
        const int c  = tile * 16 + m;
        const int cc = c < NCAND ? c : NCAND - 1;
        const int pj = perm[cc];
        int cnc;
        if (pj == H_ * K_) cnc = staloc;
        else {
            const int h = pj / K_, k = pj % K_;
            cnc = (staloc ^ (1 << h)) ^ rmask[((b * H_ + h) * K_ + k) * TP + t];
        }

        v8f accp = {}; v8f accn = {};
        #pragma unroll
        for (int chunk = 0; chunk < 2; ++chunk) {
            v16h ap, aneg;
            #pragma unroll
            for (int e = 0; e < 16; ++e) {
                const int v = e >> 1, sub = e & 1;
                const int k = (v < 4 ? 2 * v + sub : 16 + 2 * (v - 4) + sub) + hi * 8;
                const int d = chunk * 32 + k;
                // pos: accumulate log2(1+aa^2); ln2 applied once after WMMA sum
                const float dc = lutdist(cnc ^ s_pos[d][t]);
                const float aa = __builtin_fmaf(dc, s_sp[d], s_offp[d][t]);
                const float vp = fastlog2(__builtin_fmaf(aa, aa, 1.0f));
                // neg: accumulate -log2(1.1 - 1/(1+an^2))
                const float dcn = lutdist(cnc ^ s_neg[d][t]);
                const float an  = __builtin_fmaf(dcn, s_sn[d], s_offn[d][t]);
                const float p   = fastrcp(__builtin_fmaf(an, an, 1.0f));
                const float vn  = -fastlog2((1.0f + EPSV) - p);
                ap[e]   = (_Float16)vp;
                aneg[e] = (_Float16)vn;
            }
            // D[m,n] = sum_k A[m,k] * 1  (reduction over 32 d per issue)
            accp = __builtin_amdgcn_wmma_f32_16x16x32_f16(
                       false, ap,   false, bOnes, (short)0, accp, false, false);
            accn = __builtin_amdgcn_wmma_f32_16x16x32_f16(
                       false, aneg, false, bOnes, (short)0, accn, false, false);
        }

        // C/D layout: VGPR r, lanes 0-15 -> M=r ; lanes 16-31 -> M=8+r
        #pragma unroll
        for (int r = 0; r < 8; ++r) {
            const int cidx = tile * 16 + hi * 8 + r;
            const float pl = accp[r] * scale;
            const float nl = accn[r] * scale;
            float tot = pl + nl;
            tot = (cidx < NCAND) ? tot : 1e30f;
            if (tot < bestTot || (tot == bestTot && cidx < bestIdx)) {
                bestTot = tot; bestIdx = cidx; bestPos = pl; bestNeg = nl;
            }
        }
    }

    // wave32 argmin reduce (tie-break: smaller candidate index, matches argmin)
    #pragma unroll
    for (int off = 16; off >= 1; off >>= 1) {
        const float ot = __shfl_xor(bestTot, off, 32);
        const int   oi = __shfl_xor(bestIdx, off, 32);
        const float op = __shfl_xor(bestPos, off, 32);
        const float on = __shfl_xor(bestNeg, off, 32);
        if (ot < bestTot || (ot == bestTot && oi < bestIdx)) {
            bestTot = ot; bestIdx = oi; bestPos = op; bestNeg = on;
        }
    }

    if (lane == 0) {
        // recompute winning candidate location
        const int pj = perm[bestIdx];
        int sel;
        if (pj == H_ * K_) sel = staloc;
        else {
            const int h = pj / K_, k = pj % K_;
            sel = (staloc ^ (1 << h)) ^ rmask[((b * H_ + h) * K_ + k) * TP + t];
        }
        const int slot = b * TP + t;
        ws_loss[slot * 3 + 0] = bestTot;
        ws_loss[slot * 3 + 1] = bestPos;
        ws_loss[slot * 3 + 2] = bestNeg;
        ws_sel[slot] = sel;
    }
}

// d_out[3..] = (float)locations, then overwrite the BATCH selected rows
__global__ __launch_bounds__(256) void copy_loc(const int* __restrict__ loc,
                                                float* __restrict__ out)
{
    const int i = blockIdx.x * 256 + threadIdx.x;
    if (i < NNODES * TP) out[3 + i] = (float)loc[i];
}

__global__ __launch_bounds__(256) void scatter_sel(const int* __restrict__ sta_ind,
                                                   const int* __restrict__ ws_sel,
                                                   float* __restrict__ out)
{
    const int i = blockIdx.x * 256 + threadIdx.x;
    if (i < BATCH * TP) {
        const int bb = i / TP, tt = i % TP;
        out[3 + sta_ind[bb] * TP + tt] = (float)ws_sel[i];
    }
}

// deterministic fixed-order tree reduction for the three means
__global__ __launch_bounds__(256) void reduce_means(const float* __restrict__ ws_loss,
                                                    float* __restrict__ out)
{
    __shared__ float sm[256][3];
    const int tid = threadIdx.x;
    float s0 = 0.0f, s1 = 0.0f, s2 = 0.0f;
    for (int j = tid; j < BATCH * TP; j += 256) {
        s0 += ws_loss[j * 3 + 0];
        s1 += ws_loss[j * 3 + 1];
        s2 += ws_loss[j * 3 + 2];
    }
    sm[tid][0] = s0; sm[tid][1] = s1; sm[tid][2] = s2;
    __syncthreads();
    for (int off = 128; off >= 1; off >>= 1) {
        if (tid < off) {
            sm[tid][0] += sm[tid + off][0];
            sm[tid][1] += sm[tid + off][1];
            sm[tid][2] += sm[tid + off][2];
        }
        __syncthreads();
    }
    if (tid == 0) {
        const float inv = 1.0f / (float)(BATCH * TP);
        out[0] = sm[0][0] * inv;   // total loss mean
        out[1] = sm[0][1] * inv;   // pos loss mean
        out[2] = sm[0][2] * inv;   // neg loss mean
    }
}

extern "C" void kernel_launch(void* const* d_in, const int* in_sizes, int n_in,
                              void* d_out, int out_size, void* d_ws, size_t ws_size,
                              hipStream_t stream) {
    // input order: lut, sta_ind, locations, degree, pos_ind, neg_ind, random_masks, perm
    // (lut is unused: lut[x] == 32 - clz(x), computed with v_clz instead)
    const int* sta_ind   = (const int*)d_in[1];
    const int* locations = (const int*)d_in[2];
    const int* degree    = (const int*)d_in[3];
    const int* pos_ind   = (const int*)d_in[4];
    const int* neg_ind   = (const int*)d_in[5];
    const int* rmask     = (const int*)d_in[6];
    const int* perm      = (const int*)d_in[7];
    float* out = (float*)d_out;

    float* ws_loss = (float*)d_ws;                                   // 1024*3 floats
    int*   ws_sel  = (int*)((char*)d_ws + BATCH * TP * 3 * sizeof(float)); // 1024 ints

    critigraph_main<<<BATCH, 256, 0, stream>>>(sta_ind, locations, degree,
                                               pos_ind, neg_ind, rmask, perm,
                                               ws_loss, ws_sel);
    copy_loc<<<(NNODES * TP + 255) / 256, 256, 0, stream>>>(locations, out);
    scatter_sel<<<(BATCH * TP + 255) / 256, 256, 0, stream>>>(sta_ind, ws_sel, out);
    reduce_means<<<1, 256, 0, stream>>>(ws_loss, out);
}
